// MambaLayer_13030930776783
// MI455X (gfx1250) — compile-verified, matrix operand order checked by reading, not run on hardware
//
#include <hip/hip_runtime.h>

// ---------------------------------------------------------------------------
// MI455X (gfx1250) Mamba block. wave32, WMMA f32_16x16x32_f16 for all GEMMs.
// ---------------------------------------------------------------------------

typedef __attribute__((ext_vector_type(8)))  _Float16 v8h;
typedef __attribute__((ext_vector_type(16))) _Float16 v16h;
typedef __attribute__((ext_vector_type(8)))  float    v8f;

#define B_   4
#define D_   256
#define L_   4096          // 64*64
#define Ntok 16384         // B_*L_
#define DI_  512
#define DS_  16
#define DTR_ 16
#define KCONV 2304         // 9*256

// ---------------- fragment helpers ----------------------------------------
// 16-bit A-fragment (16x32): lane lm=l&15 holds row m; hi=l>>4.
// v16h elems 0..7  = K (hi*8 .. hi*8+7)       contiguous
// v16h elems 8..15 = K (16+hi*8 .. 16+hi*8+7) contiguous
// B loaded identically from Bt stored [N][K] (column-major B).
__device__ __forceinline__ v16h frag_load(const _Float16* rowk0, int hi) {
    v8h lo = *(const v8h*)(rowk0 + hi * 8);
    v8h hb = *(const v8h*)(rowk0 + 16 + hi * 8);
    return __builtin_shufflevector(lo, hb, 0,1,2,3,4,5,6,7,8,9,10,11,12,13,14,15);
}

__device__ __forceinline__ float silu_f(float x) { return x / (1.0f + __expf(-x)); }

// ---------------- weight prep: f32 -> f16, transposed to [N][K] ------------
__global__ void k_prep_weights(const float* __restrict__ W_in,   // [256][1024]
                               const float* __restrict__ W_xp,   // [512][48]
                               const float* __restrict__ W_out,  // [512][256]
                               const float* __restrict__ W_extra,// [512][256]
                               const float* __restrict__ conv1w, // [256][256][3][3]
                               _Float16* __restrict__ Wt_in,     // [1024][256]
                               _Float16* __restrict__ Wt_xp,     // [48][512]
                               _Float16* __restrict__ Wt_cat,    // [512][512]
                               _Float16* __restrict__ Wt_conv,   // [256][2304]
                               _Float16* __restrict__ zpad) {    // [512] zeros
    int idx = blockIdx.x * blockDim.x + threadIdx.x;
    if (idx < 512) zpad[idx] = (_Float16)0.0f;
    if (idx < 1024 * 256) {
        int n = idx / 256, k = idx % 256;
        Wt_in[idx] = (_Float16)W_in[k * 1024 + n];
    }
    if (idx < 48 * 512) {
        int n = idx / 512, k = idx % 512;
        Wt_xp[idx] = (_Float16)W_xp[k * 48 + n];
    }
    if (idx < 512 * 512) {
        int n = idx / 512, k = idx % 512;
        Wt_cat[idx] = (_Float16)((n < 256) ? W_out[k * 256 + n]
                                           : W_extra[k * 256 + (n - 256)]);
    }
    if (idx < 256 * KCONV) {
        int o = idx / KCONV, kk = idx % KCONV;
        int plane = kk >> 8, i = kk & 255;
        int kh = plane / 3, kw = plane % 3;
        Wt_conv[idx] = (_Float16)conv1w[((o * 256 + i) * 3 + kh) * 3 + kw];
    }
}

// ---------------- LayerNorm over D: x (B,D,H,W) -> sn_h [n][256] f16 -------
__global__ void k_layernorm(const float* __restrict__ x,
                            const float* __restrict__ g,
                            const float* __restrict__ bta,
                            _Float16* __restrict__ sn) {
    int n = blockIdx.x * blockDim.x + threadIdx.x;
    if (n >= Ntok) return;
    int b = n >> 12, l = n & (L_ - 1);
    const float* xp = x + (size_t)b * D_ * L_ + l;
    float s = 0.f, s2 = 0.f;
    for (int d = 0; d < D_; ++d) {
        float v = xp[(size_t)d * L_];
        s += v; s2 += v * v;
    }
    float mu  = s * (1.0f / D_);
    float var = s2 * (1.0f / D_) - mu * mu;
    float inv = rsqrtf(var + 1e-5f);
    _Float16* o = sn + (size_t)n * D_;
    for (int d = 0; d < D_; ++d) {
        float v = xp[(size_t)d * L_];
        o[d] = (_Float16)((v - mu) * inv * g[d] + bta[d]);
    }
}

// ---------------- templated WMMA GEMM: C[M][NN] = A * Bt^T -----------------
// block = 128 (4 waves); wave computes (MT*16) rows x (NT*16) cols.
// Branch-free inner loop: accumulators stay pinned in VGPRs.
template<int MT, int NT, int KK>
__global__ __launch_bounds__(128)
void k_gemm(const _Float16* __restrict__ A, const _Float16* __restrict__ Bt,
            float* __restrict__ C, int NN) {
    const int lane = threadIdx.x & 31;
    const int wave = threadIdx.x >> 5;
    const int hi = lane >> 4, lm = lane & 15;
    const int m0 = (blockIdx.y * 4 + wave) * (MT * 16);
    const int nBase = blockIdx.x * (NT * 16);

    v8f acc[MT][NT];
#pragma unroll
    for (int mt = 0; mt < MT; ++mt)
#pragma unroll
        for (int nt = 0; nt < NT; ++nt)
#pragma unroll
            for (int i = 0; i < 8; ++i) acc[mt][nt][i] = 0.0f;

    const _Float16* arow[MT];
#pragma unroll
    for (int mt = 0; mt < MT; ++mt)
        arow[mt] = A + (size_t)(m0 + mt * 16 + lm) * KK;
    const _Float16* brow[NT];
#pragma unroll
    for (int nt = 0; nt < NT; ++nt)
        brow[nt] = Bt + (size_t)(nBase + nt * 16 + lm) * KK;

    for (int k0 = 0; k0 < KK; k0 += 32) {
        v16h a[MT];
#pragma unroll
        for (int mt = 0; mt < MT; ++mt) a[mt] = frag_load(arow[mt] + k0, hi);
#pragma unroll
        for (int nt = 0; nt < NT; ++nt) {
            v16h bf = frag_load(brow[nt] + k0, hi);
#pragma unroll
            for (int mt = 0; mt < MT; ++mt)
                acc[mt][nt] = __builtin_amdgcn_wmma_f32_16x16x32_f16(
                    false, a[mt], false, bf, (short)0, acc[mt][nt], false, false);
        }
    }
#pragma unroll
    for (int mt = 0; mt < MT; ++mt)
#pragma unroll
        for (int nt = 0; nt < NT; ++nt) {
            int n = nBase + nt * 16 + lm;
#pragma unroll
            for (int r = 0; r < 8; ++r) {
                int m = m0 + mt * 16 + hi * 8 + r;
                C[(size_t)m * NN + n] = acc[mt][nt][r];
            }
        }
}

// ---------------- depthwise causal conv1d (DC=4) + bias + SiLU -------------
__global__ void k_dwconv_silu(const float* __restrict__ xz,
                              const float* __restrict__ conv_w, // [512][1][4]
                              const float* __restrict__ conv_b,
                              float* __restrict__ xa, _Float16* __restrict__ xah) {
    int idx = blockIdx.x * blockDim.x + threadIdx.x;
    if (idx >= Ntok * DI_) return;
    int n = idx >> 9, c = idx & 511;
    int l = n & (L_ - 1);
    float acc = conv_b[c];
    const float* w = conv_w + c * 4;
#pragma unroll
    for (int k = 0; k < 4; ++k) {
        int ls = l + k - 3;
        if (ls >= 0) acc += w[k] * xz[(size_t)(n + k - 3) * 1024 + c];
    }
    float sv = silu_f(acc);
    xa[idx] = sv;
    xah[idx] = (_Float16)sv;
}

// ---------------- fused delta + selective scan + gating --------------------
__global__ __launch_bounds__(256)
void k_scan(const float* __restrict__ proj,   // [n][48] : dt|B|C
            const float* __restrict__ xa,     // [n][512]
            const float* __restrict__ xz,     // [n][1024] (z = cols 512..)
            const float* __restrict__ W_dt,   // [16][512]
            const float* __restrict__ dt_bias,
            const float* __restrict__ A_log,  // [512][16]
            const float* __restrict__ D_skip,
            _Float16* __restrict__ yh) {      // [n][512]
    int b = blockIdx.x >> 1;
    int c = ((blockIdx.x & 1) << 8) + threadIdx.x;
    float wdt[DTR_], Ac[DS_], h[DS_];
#pragma unroll
    for (int r = 0; r < DTR_; ++r) wdt[r] = W_dt[r * DI_ + c];
#pragma unroll
    for (int s = 0; s < DS_; ++s) { Ac[s] = -__expf(A_log[c * DS_ + s]); h[s] = 0.0f; }
    float bias = dt_bias[c], dsk = D_skip[c];
    for (int l = 0; l < L_; ++l) {
        int n = (b << 12) + l;
        const float* pr = proj + (size_t)n * 48;
        float dtv = bias;
#pragma unroll
        for (int r = 0; r < DTR_; ++r) dtv += pr[r] * wdt[r];
        float delta = (dtv > 20.0f) ? dtv : log1pf(__expf(dtv));
        float xv = xa[(size_t)n * DI_ + c];
        float dbu = delta * xv;
        float y = 0.0f;
#pragma unroll
        for (int s = 0; s < DS_; ++s) {
            h[s] = h[s] * __expf(delta * Ac[s]) + dbu * pr[16 + s];
            y += h[s] * pr[32 + s];
        }
        y += xv * dsk;
        float zv = xz[(size_t)n * 1024 + 512 + c];
        y *= silu_f(zv);
        yh[(size_t)n * DI_ + c] = (_Float16)y;
    }
}

// ---------------- dual GEMM: [x_mamba | extra] = Y[16384x512] * Wcat -------
// n<256 -> xm_h NHWC f16 ; n>=256 -> d_out extra part (NCHW) = v + ah.
__global__ __launch_bounds__(128)
void k_gemm_out(const _Float16* __restrict__ Y, const _Float16* __restrict__ Wt,
                const float* __restrict__ ah,
                _Float16* __restrict__ xmh, float* __restrict__ out_extra) {
    const int KK = DI_;
    const int lane = threadIdx.x & 31;
    const int wave = threadIdx.x >> 5;
    const int hi = lane >> 4, lm = lane & 15;
    const int m0 = (blockIdx.y * 4 + wave) * 32;
    const int nBase = blockIdx.x * 64;

    v8f acc[2][4];
#pragma unroll
    for (int mt = 0; mt < 2; ++mt)
#pragma unroll
        for (int nt = 0; nt < 4; ++nt)
#pragma unroll
            for (int i = 0; i < 8; ++i) acc[mt][nt][i] = 0.0f;

    const _Float16* arow[2];
#pragma unroll
    for (int mt = 0; mt < 2; ++mt) arow[mt] = Y + (size_t)(m0 + mt * 16 + lm) * KK;
    const _Float16* brow[4];
#pragma unroll
    for (int nt = 0; nt < 4; ++nt) brow[nt] = Wt + (size_t)(nBase + nt * 16 + lm) * KK;

    for (int k0 = 0; k0 < KK; k0 += 32) {
        v16h a[2];
#pragma unroll
        for (int mt = 0; mt < 2; ++mt) a[mt] = frag_load(arow[mt] + k0, hi);
#pragma unroll
        for (int nt = 0; nt < 4; ++nt) {
            v16h bf = frag_load(brow[nt] + k0, hi);
#pragma unroll
            for (int mt = 0; mt < 2; ++mt)
                acc[mt][nt] = __builtin_amdgcn_wmma_f32_16x16x32_f16(
                    false, a[mt], false, bf, (short)0, acc[mt][nt], false, false);
        }
    }
#pragma unroll
    for (int mt = 0; mt < 2; ++mt)
#pragma unroll
        for (int nt = 0; nt < 4; ++nt) {
            int n = nBase + nt * 16 + lm;
#pragma unroll
            for (int r = 0; r < 8; ++r) {
                int m = m0 + mt * 16 + hi * 8 + r;
                float v = acc[mt][nt][r];
                if (n < 256) {
                    xmh[(size_t)m * 256 + n] = (_Float16)v;   // NHWC for conv
                } else {
                    int b = m >> 12, l = m & (L_ - 1), d = n - 256;
                    size_t o = ((size_t)(b * 256 + d) << 12) + l; // NCHW
                    out_extra[o] = v + ah[o];
                }
            }
        }
}

// ---------------- 3x3 conv as implicit WMMA GEMM + BN + ReLU6 --------------
// M=16384 pixels, N=256 out channels, K=9*256 as plane(9) x kk(256).
// Out-of-bounds rows redirect (branchless ptr select) to a zero pad buffer.
__global__ __launch_bounds__(128)
void k_conv3x3(const _Float16* __restrict__ xm,      // NHWC f16 [16384][256]
               const _Float16* __restrict__ Wt_conv, // [256][2304]
               const _Float16* __restrict__ zpad,    // [512] zeros
               const float* __restrict__ bn_g, const float* __restrict__ bn_b,
               const float* __restrict__ bn_m, const float* __restrict__ bn_v,
               float* __restrict__ out_xo) {         // NCHW f32
    const int lane = threadIdx.x & 31;
    const int wave = threadIdx.x >> 5;
    const int hi = lane >> 4, lm = lane & 15;
    const int m0 = (blockIdx.y * 4 + wave) * 32;
    const int nBase = blockIdx.x * 64;

    int bb[2], h0[2], w0[2];
#pragma unroll
    for (int mt = 0; mt < 2; ++mt) {
        int m = m0 + mt * 16 + lm;
        bb[mt] = m >> 12;
        int l = m & (L_ - 1);
        h0[mt] = l >> 6;
        w0[mt] = l & 63;
    }

    v8f acc[2][4];
#pragma unroll
    for (int mt = 0; mt < 2; ++mt)
#pragma unroll
        for (int nt = 0; nt < 4; ++nt)
#pragma unroll
            for (int i = 0; i < 8; ++i) acc[mt][nt][i] = 0.0f;

    const _Float16* brow[4];
#pragma unroll
    for (int nt = 0; nt < 4; ++nt)
        brow[nt] = Wt_conv + (size_t)(nBase + nt * 16 + lm) * KCONV;

    for (int plane = 0; plane < 9; ++plane) {
        const int dh = plane / 3 - 1, dw = plane % 3 - 1;
        const _Float16* ap[2];
#pragma unroll
        for (int mt = 0; mt < 2; ++mt) {
            int hh = h0[mt] + dh, ww = w0[mt] + dw;
            bool ok = ((unsigned)hh < 64u) && ((unsigned)ww < 64u);
            const _Float16* real =
                xm + ((size_t)((bb[mt] << 12) + (hh << 6) + ww) << 8);
            ap[mt] = ok ? real : zpad;    // branchless ptr select
        }
        const size_t bofs = (size_t)plane * 256;
#pragma unroll
        for (int kk = 0; kk < 256; kk += 32) {
            v16h a[2];
#pragma unroll
            for (int mt = 0; mt < 2; ++mt) a[mt] = frag_load(ap[mt] + kk, hi);
#pragma unroll
            for (int nt = 0; nt < 4; ++nt) {
                v16h bf = frag_load(brow[nt] + bofs + kk, hi);
#pragma unroll
                for (int mt = 0; mt < 2; ++mt)
                    acc[mt][nt] = __builtin_amdgcn_wmma_f32_16x16x32_f16(
                        false, a[mt], false, bf, (short)0, acc[mt][nt], false, false);
            }
        }
    }
#pragma unroll
    for (int nt = 0; nt < 4; ++nt) {
        int o = nBase + nt * 16 + lm;
        float sc = rsqrtf(bn_v[o] + 1e-5f) * bn_g[o];
        float sh = bn_b[o] - bn_m[o] * sc;
#pragma unroll
        for (int mt = 0; mt < 2; ++mt)
#pragma unroll
            for (int r = 0; r < 8; ++r) {
                int mm = m0 + mt * 16 + hi * 8 + r;
                int b2 = mm >> 12, ll = mm & (L_ - 1);
                float v = acc[mt][nt][r] * sc + sh;
                v = fminf(fmaxf(v, 0.0f), 6.0f);
                out_xo[((size_t)(b2 * 256 + o) << 12) + ll] = v;
            }
    }
}

// ---------------------------------------------------------------------------
extern "C" void kernel_launch(void* const* d_in, const int* in_sizes, int n_in,
                              void* d_out, int out_size, void* d_ws, size_t ws_size,
                              hipStream_t stream) {
    const float* ah      = (const float*)d_in[0];
    const float* x       = (const float*)d_in[1];
    const float* ln_g    = (const float*)d_in[2];
    const float* ln_b    = (const float*)d_in[3];
    const float* W_in    = (const float*)d_in[4];
    const float* conv_w  = (const float*)d_in[5];
    const float* conv_b  = (const float*)d_in[6];
    const float* W_xp    = (const float*)d_in[7];
    const float* W_dt    = (const float*)d_in[8];
    const float* dt_bias = (const float*)d_in[9];
    const float* A_log   = (const float*)d_in[10];
    const float* D_skip  = (const float*)d_in[11];
    const float* W_out   = (const float*)d_in[12];
    const float* W_extra = (const float*)d_in[13];
    const float* conv1w  = (const float*)d_in[14];
    const float* bn_g    = (const float*)d_in[15];
    const float* bn_b    = (const float*)d_in[16];
    const float* bn_m    = (const float*)d_in[17];
    const float* bn_v    = (const float*)d_in[18];

    float* out_xo    = (float*)d_out;                 // 4*256*4096
    float* out_extra = (float*)d_out + (size_t)B_ * D_ * L_;

    // workspace carve-up (256B aligned)
    char* ws = (char*)d_ws;
    auto carve = [&](size_t bytes) -> void* {
        void* p = (void*)ws;
        ws += (bytes + 255) & ~(size_t)255;
        return p;
    };
    _Float16* sn_h    = (_Float16*)carve((size_t)Ntok * D_ * 2);       // 8 MB
    _Float16* Wt_in   = (_Float16*)carve((size_t)1024 * 256 * 2);
    _Float16* Wt_xp   = (_Float16*)carve((size_t)48 * 512 * 2);
    _Float16* Wt_cat  = (_Float16*)carve((size_t)512 * 512 * 2);
    _Float16* Wt_conv = (_Float16*)carve((size_t)256 * KCONV * 2);
    _Float16* zpad    = (_Float16*)carve((size_t)512 * 2);
    float*    xz      = (float*)   carve((size_t)Ntok * 1024 * 4);     // 64 MB
    float*    xa      = (float*)   carve((size_t)Ntok * DI_ * 4);      // 32 MB
    _Float16* xah     = (_Float16*)carve((size_t)Ntok * DI_ * 2);      // 16 MB
    float*    proj    = (float*)   carve((size_t)Ntok * 48 * 4);       // 3 MB
    _Float16* yh      = (_Float16*)carve((size_t)Ntok * DI_ * 2);      // 16 MB
    _Float16* xmh     = (_Float16*)carve((size_t)Ntok * 256 * 2);      // 8 MB
    (void)ws_size; (void)in_sizes; (void)n_in; (void)out_size;

    // 1) weight convert/transpose (f32 -> f16 [N][K]) + zero pad buffer
    {
        int total = 256 * KCONV;   // largest range (589824)
        k_prep_weights<<<(total + 255) / 256, 256, 0, stream>>>(
            W_in, W_xp, W_out, W_extra, conv1w,
            Wt_in, Wt_xp, Wt_cat, Wt_conv, zpad);
    }
    // 2) layernorm -> sn_h
    k_layernorm<<<Ntok / 256, 256, 0, stream>>>(x, ln_g, ln_b, sn_h);
    // 3) xz = sn @ W_in   (16384 x 1024, K=256)
    k_gemm<2, 4, 256><<<dim3(1024 / 64, Ntok / 128), 128, 0, stream>>>(
        sn_h, Wt_in, xz, 1024);
    // 4) depthwise causal conv + SiLU
    k_dwconv_silu<<<(Ntok * DI_) / 256, 256, 0, stream>>>(xz, conv_w, conv_b, xa, xah);
    // 5) proj = xin @ W_xp (16384 x 48, K=512) — NT=3 exact, no masking
    k_gemm<2, 3, 512><<<dim3(1, Ntok / 128), 128, 0, stream>>>(
        xah, Wt_xp, proj, 48);
    // 6) fused delta + selective scan + D_skip + SiLU(z) gating -> yh
    k_scan<<<2 * B_, 256, 0, stream>>>(proj, xa, xz, W_dt, dt_bias, A_log, D_skip, yh);
    // 7) dual GEMM: x_mamba (NHWC f16) and extra+ah (NCHW f32, second output)
    k_gemm_out<<<dim3(512 / 64, Ntok / 128), 128, 0, stream>>>(
        yh, Wt_cat, ah, xmh, out_extra);
    // 8) 3x3 conv (implicit WMMA GEMM) + BN + ReLU6 -> first output
    k_conv3x3<<<dim3(256 / 64, Ntok / 128), 128, 0, stream>>>(
        xmh, Wt_conv, zpad, bn_g, bn_b, bn_m, bn_v, out_xo);
}
